// RNN_42898133353475
// MI455X (gfx1250) — matrix operand order repaired and checked by reading, not compiled, not run on hardware
//
#include <hip/hip_runtime.h>
#include <hip/hip_bf16.h>
#include <math.h>

// ---------------------------------------------------------------------------
// Model dims (fixed by the reference)
// ---------------------------------------------------------------------------
#define BATCH   128
#define TSEQ    256
#define EMB     300
#define EMBP    320       // K padded to a multiple of 32 (zero-filled)
#define HID     256
#define GATE    1024      // 4*HID
#define NCLS    2
#define FC1     100
#define VOCAB   30000
#define ROWS    (BATCH * TSEQ)      // 32768 flat (b*T + t) rows

typedef __bf16        v16bf  __attribute__((ext_vector_type(16)));
typedef float         v8f    __attribute__((ext_vector_type(8)));
typedef unsigned int  su32x4 __attribute__((ext_vector_type(4)));
typedef unsigned int  su32x8 __attribute__((ext_vector_type(8)));

__device__ __forceinline__ v8f wmma_bf16(v16bf a, v16bf b, v8f c) {
    // D = A(16x32 bf16) x B(32x16 bf16) + C(16x16 f32)
    return __builtin_amdgcn_wmma_f32_16x16x32_bf16(
        /*neg_a=*/false, a, /*neg_b=*/false, b,
        /*c_mod=*/(short)0, c, /*reuse_a=*/false, /*reuse_b=*/false);
}

// A-fragment (16x32 bf16): lane m = lane%16, g = lane/16.
// elems j=0..7  -> K = kbase + g*8 + j        (16B contiguous)
// elems j=8..15 -> K = kbase + 16 + g*8 + j-8 (16B contiguous)
__device__ __forceinline__ v16bf ldA_bf(const __bf16* rb, int kbase, int g) {
    v16bf r;
    ((uint4*)&r)[0] = *(const uint4*)(rb + kbase + g * 8);
    ((uint4*)&r)[1] = *(const uint4*)(rb + kbase + 16 + g * 8);
    return r;
}

// B-fragment (32x16 bf16): lane n = lane%16, g = lane/16.
// elems j=0..7  -> K = kbase + g*8 + j        (16B contiguous)
// elems j=8..15 -> K = kbase + 16 + g*8 + j-8 (16B contiguous)
__device__ __forceinline__ v16bf ldB_bf(const __bf16* rb, int kbase, int g) {
    const __bf16* p = rb + kbase + g * 8;
    v16bf r;
    ((uint4*)&r)[0] = *(const uint4*)(p);
    ((uint4*)&r)[1] = *(const uint4*)(p + 16);
    return r;
}

__device__ __forceinline__ float sigmoidf_(float x) { return 1.f / (1.f + __expf(-x)); }

// ---------------------------------------------------------------------------
// Tensor Data Mover: 1-row 2D tile, contiguous 64KB (16384 x f32) global->LDS.
// D# per CDNA5 ISA (groups 2/3 zero: dims 2..4 unused).
// ---------------------------------------------------------------------------
__device__ __forceinline__ void tdm_load_64k(unsigned long long gaddr, unsigned lds_byte) {
    su32x4 g0;
    g0[0] = 1u;                                       // count=1 (valid descriptor)
    g0[1] = lds_byte;                                 // lds_addr
    g0[2] = (unsigned)gaddr;                          // global_addr[31:0]
    g0[3] = (unsigned)(gaddr >> 32) | 0x80000000u;    // global_addr[56:32] | type=2
    su32x8 g1;
    g1[0] = 0x00020000u;   // workgroup_mask=0, data_size=2 (4B)
    g1[1] = 0x40000000u;   // tensor_dim0 = 16384 (low16 at bits 63:48)
    g1[2] = 0x00010000u;   // tensor_dim0 hi=0, tensor_dim1 = 1
    g1[3] = 0x40000000u;   // tensor_dim1 hi=0, tile_dim0 = 16384
    g1[4] = 0x00000001u;   // tile_dim1 = 1, tile_dim2 = 0
    g1[5] = 16384u;        // tensor_dim0_stride lo32
    g1[6] = 0x40000000u;   // dim0_stride hi=0, tensor_dim1_stride lo16 = 16384
    g1[7] = 0u;            // tensor_dim1_stride hi
    su32x4 gz = {0u, 0u, 0u, 0u};
    asm volatile("tensor_load_to_lds %0, %1, %2, %3"
                 :: "s"(g0), "s"(g1), "s"(gz), "s"(gz)
                 : "memory");
}

// ---------------------------------------------------------------------------
// K0a: pad+convert embedding table to bf16 (VOCAB x 320, zero-padded K)
// ---------------------------------------------------------------------------
__global__ void k_pad_emb(const float* __restrict__ emb, __bf16* __restrict__ embP) {
    long i = (long)blockIdx.x * blockDim.x + threadIdx.x;
    if (i >= (long)VOCAB * EMBP) return;
    long v = i / EMBP, e = i % EMBP;
    embP[i] = (e < EMB) ? (__bf16)emb[v * EMB + e] : (__bf16)0.f;
}

// K0b: transpose+convert a weight matrix: WT[n*Kp + k] = W[k*N + n] (bf16, K padded)
__global__ void k_packT(const float* __restrict__ W, __bf16* __restrict__ WT,
                        int K, int Kp, int N) {
    long i = (long)blockIdx.x * blockDim.x + threadIdx.x;
    if (i >= (long)N * Kp) return;
    int n = (int)(i / Kp), k = (int)(i % Kp);
    WT[i] = (k < K) ? (__bf16)W[(long)k * N + n] : (__bf16)0.f;
}

// ---------------------------------------------------------------------------
// K1: xW[t][b][:] = emb[tokens[b,t], :] @ Wi + b_lstm  (fused gather + GEMM)
// stored T-MAJOR so the LSTM's per-step block is one contiguous 64KB/WG tile.
// ---------------------------------------------------------------------------
__global__ __launch_bounds__(128) void k_embed_xw(const int* __restrict__ tokens,
                                                  const __bf16* __restrict__ embP,
                                                  const __bf16* __restrict__ WiTb,
                                                  const float* __restrict__ b_lstm,
                                                  float* __restrict__ xW) {
    int wave = blockIdx.x * 4 + (threadIdx.x >> 5);
    int lane = threadIdx.x & 31;
    int g = lane >> 4, mn = lane & 15;
    int colTile = wave & 63;           // 64 tiles of 16 over GATE
    int rowTile = wave >> 6;           // 2048 tiles of 16 over ROWS
    const __bf16* arow = embP + (long)tokens[rowTile * 16 + mn] * EMBP;
    const __bf16* brow = WiTb + (long)(colTile * 16 + mn) * EMBP;
    float bias = b_lstm[colTile * 16 + mn];
    v8f acc;
#pragma unroll
    for (int v = 0; v < 8; ++v) acc[v] = bias;
#pragma unroll
    for (int kb = 0; kb < EMBP / 32; ++kb)
        acc = wmma_bf16(ldA_bf(arow, kb * 32, g), ldB_bf(brow, kb * 32, g), acc);
#pragma unroll
    for (int v = 0; v < 8; ++v) {
        long row = (long)rowTile * 16 + v + g * 8;     // flat b*T + t
        long b = row >> 8, t = row & 255;
        xW[(t * BATCH + b) * GATE + colTile * 16 + mn] = acc[v];
    }
}

// ---------------------------------------------------------------------------
// K2: LSTM scan. One WG per 16-batch-row tile (8 WGs, no global sync needed).
// Per-step xW block (64KB contiguous) is TDM-prefetched into double-buffered
// LDS one step ahead (tensor_load_to_lds + s_wait_tensorcnt). h is a
// double-buffered bf16 LDS tile; c stays in VGPRs for all 256 steps.
// Wave w owns hidden cols [w*32,w*32+32) across ALL FOUR gates, so the
// i/f/g/o nonlinearity + cell update is pure per-lane register math.
// ---------------------------------------------------------------------------
__global__ __launch_bounds__(256) void k_lstm(const float* __restrict__ xW,
                                              const __bf16* __restrict__ WhTb,
                                              __bf16* __restrict__ hs_bf) {
    extern __shared__ char smem[];                 // 147456 B dynamic LDS
    float*  xwb = (float*)smem;                    // 2 x 16 x 1024 f32 (128KB)
    __bf16* hb  = (__bf16*)(smem + 131072);        // 2 x 16 x 256 bf16 (16KB)
    int tid = threadIdx.x, lane = tid & 31, wv = tid >> 5;   // wave 0..7
    int btile = blockIdx.x;                                  // 0..7
    int g = lane >> 4, mn = lane & 15;

    for (int i = tid; i < 16 * HID; i += 256) hb[i] = (__bf16)0.f;   // h(0) = 0

    float c[2][8];
#pragma unroll
    for (int ct = 0; ct < 2; ++ct)
#pragma unroll
        for (int v = 0; v < 8; ++v) c[ct][v] = 0.f;

    const __bf16* brow[4][2];
#pragma unroll
    for (int G = 0; G < 4; ++G)
#pragma unroll
        for (int ct = 0; ct < 2; ++ct)
            brow[G][ct] = WhTb + (long)(G * HID + wv * 32 + ct * 16 + mn) * HID;

    unsigned long long gbase =
        (unsigned long long)(uintptr_t)xW + (unsigned long long)btile * 16 * GATE * 4;
    unsigned ldsXw = (unsigned)(uintptr_t)xwb;     // LDS byte offset = addr[31:0]

    if (wv == 0) tdm_load_64k(gbase, ldsXw);       // prefetch step 0

    for (int t = 0; t < TSEQ; ++t) {
        int cur = t & 1, nxt = cur ^ 1;
        if (wv == 0) __builtin_amdgcn_s_wait_tensorcnt(0);  // xwb[cur] landed
        __syncthreads();                                    // + hb[cur] ready
        if (wv == 0 && t + 1 < TSEQ)
            tdm_load_64k(gbase + (unsigned long long)(t + 1) * BATCH * GATE * 4,
                         ldsXw + (unsigned)nxt * 65536u);

        const float* xc = xwb + (size_t)cur * 16 * 1024;
        v8f acc[4][2];
#pragma unroll
        for (int G = 0; G < 4; ++G)
#pragma unroll
            for (int ct = 0; ct < 2; ++ct) {
                int col = G * HID + wv * 32 + ct * 16 + mn;
#pragma unroll
                for (int v = 0; v < 8; ++v)
                    acc[G][ct][v] = xc[(v + g * 8) * GATE + col];
            }
        const __bf16* hr = hb + (size_t)cur * 16 * HID + (size_t)mn * HID;
        for (int kb = 0; kb < 8; ++kb) {
            v16bf a;   // two 16B runs -> 2x ds_load_b128
            ((uint4*)&a)[0] = *(const uint4*)(hr + kb * 32 + g * 8);
            ((uint4*)&a)[1] = *(const uint4*)(hr + kb * 32 + 16 + g * 8);
#pragma unroll
            for (int G = 0; G < 4; ++G)
#pragma unroll
                for (int ct = 0; ct < 2; ++ct)
                    acc[G][ct] = wmma_bf16(a, ldB_bf(brow[G][ct], kb * 32, g), acc[G][ct]);
        }
#pragma unroll
        for (int ct = 0; ct < 2; ++ct)
#pragma unroll
            for (int v = 0; v < 8; ++v) {
                float iv = sigmoidf_(acc[0][ct][v]);
                float fv = sigmoidf_(acc[1][ct][v]);
                float gv = tanhf(acc[2][ct][v]);
                float ov = sigmoidf_(acc[3][ct][v]);
                float cc = fv * c[ct][v] + iv * gv;
                c[ct][v] = cc;
                float h = ov * tanhf(cc);
                int mrow = v + g * 8;
                int hcol = wv * 32 + ct * 16 + mn;
                hb[(size_t)nxt * 16 * HID + mrow * HID + hcol] = (__bf16)h;
                hs_bf[((long)(btile * 16 + mrow) * TSEQ + t) * HID + hcol] = (__bf16)h;
            }
    }
}

// ---------------------------------------------------------------------------
// K3: q,k,v projections (one A-fragment feeds 3 WMMAs). q scaled 1/sqrt(H);
// all outputs stored bf16; v stored TRANSPOSED per batch for the ctx GEMM.
// ---------------------------------------------------------------------------
__global__ __launch_bounds__(128) void k_qkv(const __bf16* __restrict__ hs_bf,
                                             const __bf16* __restrict__ WqTb,
                                             const __bf16* __restrict__ WkTb,
                                             const __bf16* __restrict__ WvTb,
                                             const float* __restrict__ bq,
                                             const float* __restrict__ bk,
                                             const float* __restrict__ bv,
                                             __bf16* __restrict__ q_bf,
                                             __bf16* __restrict__ k_bf,
                                             __bf16* __restrict__ vT_bf) {
    int wave = blockIdx.x * 4 + (threadIdx.x >> 5);
    int lane = threadIdx.x & 31;
    int g = lane >> 4, mn = lane & 15;
    int colTile = wave & 15;            // 16 tiles over HID
    int rowTile = wave >> 4;            // 2048 tiles over ROWS
    int col = colTile * 16 + mn;
    const __bf16* arow = hs_bf + (long)(rowTile * 16 + mn) * HID;
    const __bf16* bqr = WqTb + (long)col * HID;
    const __bf16* bkr = WkTb + (long)col * HID;
    const __bf16* bvr = WvTb + (long)col * HID;
    v8f aq, ak, av;
#pragma unroll
    for (int v = 0; v < 8; ++v) { aq[v] = bq[col]; ak[v] = bk[col]; av[v] = bv[col]; }
#pragma unroll
    for (int kb = 0; kb < 8; ++kb) {
        v16bf a = ldA_bf(arow, kb * 32, g);
        aq = wmma_bf16(a, ldB_bf(bqr, kb * 32, g), aq);
        ak = wmma_bf16(a, ldB_bf(bkr, kb * 32, g), ak);
        av = wmma_bf16(a, ldB_bf(bvr, kb * 32, g), av);
    }
#pragma unroll
    for (int v = 0; v < 8; ++v) {
        long row = (long)rowTile * 16 + v + g * 8;      // flat b*T + t
        long b = row >> 8, t = row & 255;
        q_bf[row * HID + col] = (__bf16)(aq[v] * 0.0625f);   // / sqrt(256)
        k_bf[row * HID + col] = (__bf16)ak[v];
        vT_bf[(b * HID + col) * TSEQ + t] = (__bf16)av[v];
    }
}

// ---------------------------------------------------------------------------
// K4: scores[b,qr,kr] = dot(q[b,qr,:], k[b,kr,:]); query-row mask applied.
// ---------------------------------------------------------------------------
__global__ __launch_bounds__(128) void k_scores(const __bf16* __restrict__ q_bf,
                                                const __bf16* __restrict__ k_bf,
                                                const unsigned char* __restrict__ mask,
                                                float* __restrict__ sc) {
    int wave = blockIdx.x * 4 + (threadIdx.x >> 5);
    int lane = threadIdx.x & 31;
    int g = lane >> 4, mn = lane & 15;
    int kt = wave & 15, qt = (wave >> 4) & 15, b = wave >> 8;
    const __bf16* arow = q_bf + (long)(b * TSEQ + qt * 16 + mn) * HID;
    const __bf16* brow = k_bf + (long)(b * TSEQ + kt * 16 + mn) * HID;
    v8f acc = {};
#pragma unroll
    for (int kb = 0; kb < 8; ++kb)
        acc = wmma_bf16(ldA_bf(arow, kb * 32, g), ldB_bf(brow, kb * 32, g), acc);
#pragma unroll
    for (int v = 0; v < 8; ++v) {
        int qr = qt * 16 + v + g * 8;
        float val = mask[b * TSEQ + qr] ? acc[v] : -3.4028235e38f;
        sc[((long)b * TSEQ + qr) * TSEQ + kt * 16 + mn] = val;
    }
}

// K5: row softmax over 256 keys; weights written bf16 (A-operand of ctx GEMM)
__global__ __launch_bounds__(256) void k_softmax(const float* __restrict__ sc,
                                                 __bf16* __restrict__ wgt) {
    __shared__ float red[256];
    long row = blockIdx.x;
    int tid = threadIdx.x;
    float x = sc[row * TSEQ + tid];
    red[tid] = x; __syncthreads();
    for (int s = 128; s > 0; s >>= 1) { if (tid < s) red[tid] = fmaxf(red[tid], red[tid + s]); __syncthreads(); }
    float m = red[0]; __syncthreads();
    float e = __expf(x - m);
    red[tid] = e; __syncthreads();
    for (int s = 128; s > 0; s >>= 1) { if (tid < s) red[tid] += red[tid + s]; __syncthreads(); }
    wgt[row * TSEQ + tid] = (__bf16)(e / red[0]);
}

// K6a: ctx = softmax_weights @ v  (per batch; vT gives contiguous B K-runs)
__global__ __launch_bounds__(128) void k_ctx(const __bf16* __restrict__ wgt,
                                             const __bf16* __restrict__ vT,
                                             __bf16* __restrict__ ctx) {
    int wave = blockIdx.x * 4 + (threadIdx.x >> 5);
    int lane = threadIdx.x & 31;
    int g = lane >> 4, mn = lane & 15;
    int dt = wave & 15, qt = (wave >> 4) & 15, b = wave >> 8;
    const __bf16* arow = wgt + (long)(b * TSEQ + qt * 16 + mn) * TSEQ;
    const __bf16* brow = vT + ((long)b * HID + dt * 16 + mn) * TSEQ;
    v8f acc = {};
#pragma unroll
    for (int kb = 0; kb < 8; ++kb)
        acc = wmma_bf16(ldA_bf(arow, kb * 32, g), ldB_bf(brow, kb * 32, g), acc);
#pragma unroll
    for (int v = 0; v < 8; ++v)
        ctx[((long)b * TSEQ + qt * 16 + v + g * 8) * HID + dt * 16 + mn] = (__bf16)acc[v];
}

// K6b: attn = ctx @ Wo + bo  (fp32 out; memory layout == flat (B, T*H))
__global__ __launch_bounds__(128) void k_proj(const __bf16* __restrict__ ctx,
                                              const __bf16* __restrict__ WoTb,
                                              const float* __restrict__ bo,
                                              float* __restrict__ attnp) {
    int wave = blockIdx.x * 4 + (threadIdx.x >> 5);
    int lane = threadIdx.x & 31;
    int g = lane >> 4, mn = lane & 15;
    int colTile = wave & 15, rowTile = wave >> 4;
    int col = colTile * 16 + mn;
    const __bf16* arow = ctx + (long)(rowTile * 16 + mn) * HID;
    const __bf16* brow = WoTb + (long)col * HID;
    v8f acc;
#pragma unroll
    for (int v = 0; v < 8; ++v) acc[v] = bo[col];
#pragma unroll
    for (int kb = 0; kb < 8; ++kb)
        acc = wmma_bf16(ldA_bf(arow, kb * 32, g), ldB_bf(brow, kb * 32, g), acc);
#pragma unroll
    for (int v = 0; v < 8; ++v)
        attnp[(long)(rowTile * 16 + v + g * 8) * HID + col] = acc[v];
}

// K7: h1 = relu(flat @ W1 + b1)   (K = 65536 reduction per output, 128x100)
__global__ __launch_bounds__(256) void k_fc1(const float* __restrict__ flat,
                                             const float* __restrict__ W1,
                                             const float* __restrict__ b1,
                                             float* __restrict__ h1) {
    __shared__ float red[256];
    int b = blockIdx.x / FC1, n = blockIdx.x % FC1;
    int tid = threadIdx.x;
    const float* fb = flat + (long)b * (TSEQ * HID);
    float s = 0.f;
    for (int i = tid; i < TSEQ * HID; i += 256) s += fb[i] * W1[(long)i * FC1 + n];
    red[tid] = s; __syncthreads();
    for (int st = 128; st > 0; st >>= 1) { if (tid < st) red[tid] += red[tid + st]; __syncthreads(); }
    if (tid == 0) h1[b * FC1 + n] = fmaxf(red[0] + b1[n], 0.f);
}

// K8: logits + 2-class softmax
__global__ __launch_bounds__(128) void k_logits(const float* __restrict__ h1,
                                                const float* __restrict__ W2,
                                                const float* __restrict__ b2,
                                                float* __restrict__ out) {
    int b = threadIdx.x;
    if (b >= BATCH) return;
    float l0 = b2[0], l1 = b2[1];
    for (int j = 0; j < FC1; ++j) {
        float h = h1[b * FC1 + j];
        l0 += h * W2[j * NCLS + 0];
        l1 += h * W2[j * NCLS + 1];
    }
    float m = fmaxf(l0, l1);
    float e0 = __expf(l0 - m), e1 = __expf(l1 - m);
    float inv = 1.f / (e0 + e1);
    out[b * NCLS + 0] = e0 * inv;
    out[b * NCLS + 1] = e1 * inv;
}

// ---------------------------------------------------------------------------
// Orchestration
// ---------------------------------------------------------------------------
extern "C" void kernel_launch(void* const* d_in, const int* in_sizes, int n_in,
                              void* d_out, int out_size, void* d_ws, size_t ws_size,
                              hipStream_t stream) {
    (void)in_sizes; (void)n_in; (void)out_size; (void)ws_size;
    const int*   tokens = (const int*)d_in[0];
    const unsigned char* mask = (const unsigned char*)d_in[1];  // jax bool -> 1 byte
    const float* emb  = (const float*)d_in[2];
    const float* Wi   = (const float*)d_in[3];
    const float* Wh   = (const float*)d_in[4];
    const float* blm  = (const float*)d_in[5];
    const float* Wq   = (const float*)d_in[6];  const float* bq = (const float*)d_in[7];
    const float* Wk   = (const float*)d_in[8];  const float* bk = (const float*)d_in[9];
    const float* Wv   = (const float*)d_in[10]; const float* bv = (const float*)d_in[11];
    const float* Wo   = (const float*)d_in[12]; const float* bo = (const float*)d_in[13];
    const float* W1   = (const float*)d_in[14]; const float* b1 = (const float*)d_in[15];
    const float* W2   = (const float*)d_in[16]; const float* b2 = (const float*)d_in[17];

    // ---- workspace layout (bytes; all 16B-aligned) ----
    char* wsb = (char*)d_ws;
    __bf16* embP  = (__bf16*)wsb;                       // 30000*320*2 = 19,200,000
    __bf16* WiTb  = (__bf16*)(wsb + 19200000);          // 1024*320*2  =    655,360
    __bf16* WhTb  = (__bf16*)(wsb + 19855360);          // 1024*256*2  =    524,288
    __bf16* WqTb  = (__bf16*)(wsb + 20379648);          // 256*256*2   =    131,072
    __bf16* WkTb  = (__bf16*)(wsb + 20510720);
    __bf16* WvTb  = (__bf16*)(wsb + 20641792);
    __bf16* WoTb  = (__bf16*)(wsb + 20772864);
    __bf16* hs_bf = (__bf16*)(wsb + 20903936);          // 32768*256*2 = 16,777,216
    char*   big   = wsb + 37681152;                     // 134,217,728-byte region
    float*  xW    = (float*)big;                        // t-major, dead after LSTM
    __bf16* q_bf  = (__bf16*)(big + 0);                 // 16 MiB each
    __bf16* k_bf  = (__bf16*)(big + 16777216);
    __bf16* vT_bf = (__bf16*)(big + 33554432);
    float*  sc    = (float*) (big + 50331648);          // 32 MiB fp32 scores
    __bf16* wgtbf = (__bf16*)(big + 83886080);
    __bf16* ctxbf = (__bf16*)(big + 0);                 // reuse: q dead after scores
    float*  attnp = (float*) (big + 16777216);          // reuse: k,vT dead after ctx
    float*  h1    = (float*) (big + 50331648);          // reuse: sc dead after softmax
    float*  out   = (float*)d_out;

    // K0: pack operands (bf16, transposed, K zero-padded)
    k_pad_emb<<<(VOCAB * EMBP + 255) / 256, 256, 0, stream>>>(emb, embP);
    k_packT  <<<(GATE * EMBP + 255) / 256, 256, 0, stream>>>(Wi, WiTb, EMB, EMBP, GATE);
    k_packT  <<<(GATE * HID  + 255) / 256, 256, 0, stream>>>(Wh, WhTb, HID, HID, GATE);
    k_packT  <<<(HID * HID   + 255) / 256, 256, 0, stream>>>(Wq, WqTb, HID, HID, HID);
    k_packT  <<<(HID * HID   + 255) / 256, 256, 0, stream>>>(Wk, WkTb, HID, HID, HID);
    k_packT  <<<(HID * HID   + 255) / 256, 256, 0, stream>>>(Wv, WvTb, HID, HID, HID);
    k_packT  <<<(HID * HID   + 255) / 256, 256, 0, stream>>>(Wo, WoTb, HID, HID, HID);

    k_embed_xw<<<32768, 128, 0, stream>>>(tokens, embP, WiTb, blm, xW);
    k_lstm    <<<8, 256, 147456, stream>>>(xW, WhTb, hs_bf);
    k_qkv     <<<8192, 128, 0, stream>>>(hs_bf, WqTb, WkTb, WvTb, bq, bk, bv,
                                         q_bf, k_bf, vT_bf);
    k_scores  <<<8192, 128, 0, stream>>>(q_bf, k_bf, mask, sc);
    k_softmax <<<32768, 256, 0, stream>>>(sc, wgtbf);
    k_ctx     <<<8192, 128, 0, stream>>>(wgtbf, vT_bf, ctxbf);
    k_proj    <<<8192, 128, 0, stream>>>(ctxbf, WoTb, bo, attnp);
    k_fc1     <<<BATCH * FC1, 256, 0, stream>>>(attnp, W1, b1, h1);
    k_logits  <<<1, 128, 0, stream>>>(h1, W2, b2, out);
}
